// MIM_Model_31086973288645
// MI455X (gfx1250) — matrix operand order repaired and checked by reading, not compiled, not run on hardware
//
#include <hip/hip_runtime.h>
#include <cstddef>
#include <utility>

// ---------------------------------------------------------------------------
// MIM (PredRNN-MIM) forward for MI455X / gfx1250.
// Convs lowered to implicit GEMM: explicit fp16 im2col (lives in 192MB L2)
// feeding v_wmma_f32_16x16x32_f16 (f32 accumulate), wave32, 8 waves/block.
// GEMM: 32x16 tile per wave (2 WMMA/k-step, shared B frag), 32x32 A tile
// double-buffered in LDS (1 barrier/k-step). ~154 GFLOP/timestep.
// ---------------------------------------------------------------------------

typedef __attribute__((ext_vector_type(16))) _Float16 v16h;
typedef __attribute__((ext_vector_type(8)))  _Float16 h8;
typedef __attribute__((ext_vector_type(8)))  float    v8f;

#define HW   1024          // 32*32
#define NB   8             // batch
#define NHC  64            // num_hidden
#define NPIX (NB * HW)     // GEMM N = 8192
static constexpr int TT = 20, STEPS = 19, FCH = 16, INLEN = 10;

__device__ __forceinline__ float sigf(float x) { return 1.f / (1.f + __expf(-x)); }

// ---------------------------- utility kernels ------------------------------

__global__ void k_zero(float* __restrict__ p, int n) {
    int i = blockIdx.x * blockDim.x + threadIdx.x;
    if (i < n) p[i] = 0.f;
}

__global__ void k_sub(const float* __restrict__ a, const float* __restrict__ b,
                      float* __restrict__ o, int n) {
    int i = blockIdx.x * blockDim.x + threadIdx.x;
    if (i < n) o[i] = a[i] - b[i];
}

__global__ void k_mul_tanh(const float* __restrict__ o, const float* __restrict__ z,
                           float* __restrict__ h, int n) {
    int i = blockIdx.x * blockDim.x + threadIdx.x;
    if (i < n) h[i] = o[i] * tanhf(z[i]);
}

// fp32 conv weight [Cout][K] -> fp16 GEMM A [Cout][Kpad] (zero-padded K)
__global__ void k_wcast(const float* __restrict__ W, _Float16* __restrict__ A,
                        int K, int Kpad, int total) {
    int i = blockIdx.x * blockDim.x + threadIdx.x;
    if (i >= total) return;
    int kp = i % Kpad, co = i / Kpad;
    A[i] = (kp < K) ? (_Float16)W[(size_t)co * K + kp] : (_Float16)0.f;
}

// im2col for 5x5/pad2 conv: X fp32 [B][Cin][32][32] -> Bt fp16 [NPIX][Kpad]
__global__ void k_im2col(const float* __restrict__ X, _Float16* __restrict__ Bt,
                         int Cin, int K, int Kpad) {
    size_t i = (size_t)blockIdx.x * blockDim.x + threadIdx.x;
    if (i >= (size_t)NPIX * Kpad) return;
    int kp = (int)(i % Kpad);
    int n  = (int)(i / Kpad);
    float v = 0.f;
    if (kp < K) {
        int ci = kp / 25, r = kp % 25, kh = r / 5, kw = r % 5;
        int b = n >> 10, hw = n & 1023, oh = hw >> 5, ow = hw & 31;
        int ih = oh + kh - 2, iw = ow + kw - 2;
        if ((unsigned)ih < 32u && (unsigned)iw < 32u)
            v = X[((size_t)b * Cin + ci) * HW + ih * 32 + iw];
    }
    Bt[i] = (_Float16)v;
}

// transpose for 1x1 conv: X fp32 [B][Cin][HW] -> Bt fp16 [NPIX][Cin]
__global__ void k_xpose(const float* __restrict__ X, _Float16* __restrict__ Bt, int Cin) {
    size_t i = (size_t)blockIdx.x * blockDim.x + threadIdx.x;
    if (i >= (size_t)NPIX * Cin) return;
    int ci = (int)(i % Cin);
    int n  = (int)(i / Cin);
    int b = n >> 10, hw = n & 1023;
    Bt[i] = (_Float16)X[((size_t)b * Cin + ci) * HW + hw];
}

// ------------------------------ WMMA GEMM ----------------------------------
// C[M,N] = A[M,Kpad] * Bt^T.  A fp16 row-major [M][Kpad]; Bt fp16 [N][Kpad]
// (column of B contiguous in K).  Block = 8 waves, macro tile M=32, N=128:
// wave w owns N subtile w*16 and both M subtiles (2 WMMA/k-step, one shared
// B fragment).  32x32 A tile double-buffered in LDS -> 1 barrier per k-step.
// M=16 matrices (final conv) handled by a wave-uniform guard (EXEC stays ~0,
// per WMMA restrictions).  Output written as fp32 [B][Cout][HW].
__global__ __launch_bounds__(256) void k_gemm(const _Float16* __restrict__ A,
                                              const _Float16* __restrict__ Bt,
                                              float* __restrict__ out,
                                              int Kpad, int Cout) {
    __shared__ __align__(16) _Float16 As[2][32 * 32];
    const int tid  = threadIdx.x;
    const int wave = tid >> 5;
    const int lane = tid & 31;
    const int mt   = blockIdx.y * 32;                  // M tile base (Cout)
    const int nt   = blockIdx.x * 128 + wave * 16;     // N tile base (pixel)
    const int mrow = lane & 15;
    const int kgrp = lane >> 4;                        // 0 | 1
    const int kb   = kgrp * 8;
    const bool hasM1 = (mt + 16) < Cout;               // wave-uniform

    // stage 32x32 fp16 A tile (2KB) into LDS buffer p: threads 0..127
    auto stage = [&](int p, int k0) {
        if (tid < 128) {
            const int r = tid >> 2, ccol = (tid & 3) * 8;
            h8 v;
#pragma unroll
            for (int j = 0; j < 8; ++j) v[j] = (_Float16)0.f;
            if (mt + r < Cout)
                v = *(const h8*)&A[(size_t)(mt + r) * Kpad + k0 + ccol];
            *(h8*)&As[p][r * 32 + ccol] = v;
        }
    };

    v8f acc0 = {}, acc1 = {};
    const int nkt = Kpad >> 5;
    stage(0, 0);
    __syncthreads();

    for (int kt = 0; kt < nkt; ++kt) {
        const int p = kt & 1;
        if (kt + 1 < nkt) stage(p ^ 1, (kt + 1) * 32);   // fill other buffer

        // B fragment: col n = nt+mrow; lanes 0-15 K{0..7,16..23}, 16-31 K{8..15,24..31}
        const _Float16* bp = &Bt[(size_t)(nt + mrow) * Kpad + kt * 32 + kb];
        __builtin_prefetch(bp + 32, 0, 1);               // next k-chunk -> global_prefetch
        h8 b0 = *(const h8*)bp;
        h8 b1 = *(const h8*)(bp + 16);
        v16h b = __builtin_shufflevector(b0, b1, 0,1,2,3,4,5,6,7,8,9,10,11,12,13,14,15);

        // A fragments (ISA 16-bit A 16x32 layout): lanes 0-15 K{0..7,16..23},
        // lanes 16-31 K{8..15,24..31}
        h8 a00 = *(const h8*)&As[p][mrow * 32 + kb];
        h8 a01 = *(const h8*)&As[p][mrow * 32 + 16 + kb];
        v16h a0 = __builtin_shufflevector(a00, a01, 0,1,2,3,4,5,6,7,8,9,10,11,12,13,14,15);
        acc0 = __builtin_amdgcn_wmma_f32_16x16x32_f16(
                   false, a0, false, b, (short)0, acc0, false, false);
        if (hasM1) {
            h8 a10 = *(const h8*)&As[p][(16 + mrow) * 32 + kb];
            h8 a11 = *(const h8*)&As[p][(16 + mrow) * 32 + 16 + kb];
            v16h a1 = __builtin_shufflevector(a10, a11, 0,1,2,3,4,5,6,7,8,9,10,11,12,13,14,15);
            acc1 = __builtin_amdgcn_wmma_f32_16x16x32_f16(
                       false, a1, false, b, (short)0, acc1, false, false);
        }
        __syncthreads();   // publish next A buffer / retire current reads
    }

    // C/D layout: VGPR r -> M = r + 8*kgrp, N = lane&15
    const int n    = nt + mrow;
    const int bidx = n >> 10, hw = n & 1023;
#pragma unroll
    for (int r = 0; r < 8; ++r) {
        const int m = r + 8 * kgrp;
        out[((size_t)bidx * Cout + (mt + m)) * HW + hw] = acc0[r];
    }
    if (hasM1) {
#pragma unroll
        for (int r = 0; r < 8; ++r) {
            const int m = 16 + r + 8 * kgrp;
            out[((size_t)bidx * Cout + (mt + m)) * HW + hw] = acc1[r];
        }
    }
}

// ------------------------------ LayerNorm ----------------------------------
// nn.LayerNorm([C,H,W]) per sample; one block per batch element.
__global__ __launch_bounds__(512) void k_layernorm(const float* __restrict__ X,
                                                   const float* __restrict__ g,
                                                   const float* __restrict__ bb,
                                                   float* __restrict__ out, int C) {
    __shared__ float s1[512], s2[512];
    const int b = blockIdx.x;
    const size_t ne = (size_t)C * HW;
    const float* xb = X + (size_t)b * ne;
    float sum = 0.f, sq = 0.f;
    for (size_t i = threadIdx.x; i < ne; i += 512) { float v = xb[i]; sum += v; sq += v * v; }
    s1[threadIdx.x] = sum; s2[threadIdx.x] = sq;
    __syncthreads();
    for (int off = 256; off > 0; off >>= 1) {
        if ((int)threadIdx.x < off) { s1[threadIdx.x] += s1[threadIdx.x + off];
                                      s2[threadIdx.x] += s2[threadIdx.x + off]; }
        __syncthreads();
    }
    const float mu = s1[0] / (float)ne;
    const float rs = rsqrtf(s2[0] / (float)ne - mu * mu + 1e-5f);
    float* ob = out + (size_t)b * ne;
    for (size_t i = threadIdx.x; i < ne; i += 512)
        ob[i] = (xb[i] - mu) * rs * g[i] + bb[i];
}

// ---------------------------- fused gate math ------------------------------

#define ELEM_IDX()                                            \
    int i = blockIdx.x * blockDim.x + threadIdx.x;            \
    if (i >= NB * NHC * HW) return;                           \
    const int hw = i % HW;                                    \
    const int ch = (i / HW) % NHC;                            \
    const int b  = i / (HW * NHC);                            \
    const size_t S = (size_t)NHC * HW; (void)hw; (void)S;

// ST-LSTM: c/m update + concat memory
__global__ void k_stlstm_gates(const float* __restrict__ xcc, const float* __restrict__ hcc,
                               const float* __restrict__ mcc, const float* __restrict__ c,
                               const float* __restrict__ m, float* __restrict__ c_new,
                               float* __restrict__ m_new, float* __restrict__ mem) {
    ELEM_IDX();
    const size_t xo = ((size_t)b * 7 * NHC + ch) * HW + hw;
    const size_t ho = ((size_t)b * 4 * NHC + ch) * HW + hw;
    const size_t mo = ((size_t)b * 3 * NHC + ch) * HW + hw;
    float i_t = sigf(xcc[xo]         + hcc[ho]);
    float f_t = sigf(xcc[xo + S]     + hcc[ho + S] + 1.f);
    float g_t = tanhf(xcc[xo + 2*S]  + hcc[ho + 2*S]);
    float cn  = f_t * c[i] + i_t * g_t;
    float i_p = sigf(xcc[xo + 3*S]   + mcc[mo]);
    float f_p = sigf(xcc[xo + 4*S]   + mcc[mo + S] + 1.f);
    float g_p = tanhf(xcc[xo + 5*S]  + mcc[mo + 2*S]);
    float mn  = f_p * m[i] + i_p * g_p;
    c_new[i] = cn; m_new[i] = mn;
    const size_t co = ((size_t)b * 2 * NHC + ch) * HW + hw;
    mem[co] = cn; mem[co + S] = mn;
}

// ST-LSTM output gate: h = sig(o_x + o_h + o_cc) * tanh(z)
__global__ void k_stlstm_out(const float* __restrict__ xcc, const float* __restrict__ hcc,
                             const float* __restrict__ occ, const float* __restrict__ z,
                             float* __restrict__ h) {
    ELEM_IDX();
    const float ox = xcc[((size_t)b * 7 * NHC + 6 * NHC + ch) * HW + hw];
    const float oh = hcc[((size_t)b * 4 * NHC + 3 * NHC + ch) * HW + hw];
    h[i] = sigf(ox + oh + occ[i]) * tanhf(z[i]);
}

// MIMN / MIM-S gating (identical structure, gate order i,g,f,o)
__global__ void k_mimn(const float* __restrict__ hcc, const float* __restrict__ xcc,
                       const float* __restrict__ c, const float* __restrict__ ctw,
                       const float* __restrict__ ocw, float* __restrict__ h_new,
                       float* __restrict__ c_new) {
    ELEM_IDX();
    const size_t ho = ((size_t)b * 4 * NHC + ch) * HW + hw;
    const size_t xo = ho;  // same [B][4NH][HW] layout
    const float cw_i = ctw[(size_t)ch * HW + hw];
    const float cw_f = ctw[((size_t)NHC + ch) * HW + hw];
    const float cv = c[i];
    float i_ = sigf(hcc[ho]         + cv * cw_i + xcc[xo]);
    float f_ = sigf(hcc[ho + 2*S]   + cv * cw_f + xcc[xo + 2*S] + 1.f);
    float g_ = tanhf(hcc[ho + S]    + xcc[xo + S]);
    float o_ = hcc[ho + 3*S] + xcc[xo + 3*S];
    float cn = f_ * cv + i_ * g_;
    c_new[i] = cn;
    h_new[i] = sigf(o_ + cn * ocw[(size_t)ch * HW + hw]) * tanhf(cn);
}

// MIMBlock gating: tcc(i,g,o), scc(i,g,f,o), xcc(i,g,f,o), m, c2(from MIM-S)
__global__ void k_mimblock_gates(const float* __restrict__ tcc, const float* __restrict__ scc,
                                 const float* __restrict__ xcc, const float* __restrict__ m,
                                 const float* __restrict__ c2, float* __restrict__ new_c,
                                 float* __restrict__ new_m, float* __restrict__ o_out,
                                 float* __restrict__ cell) {
    ELEM_IDX();
    const size_t to = ((size_t)b * 3 * NHC + ch) * HW + hw;
    const size_t so = ((size_t)b * 4 * NHC + ch) * HW + hw;
    const size_t xo = so;
    float i_x = xcc[xo], g_x = xcc[xo + S], f_x = xcc[xo + 2*S], o_x = xcc[xo + 3*S];
    float ii  = sigf(i_x + tcc[to]);
    float i_  = sigf(i_x + scc[so]);
    float gg  = tanhf(g_x + tcc[to + S]);
    float g_  = tanhf(g_x + scc[so + S]);
    float f_  = sigf(f_x + scc[so + 2*S] + 1.f);
    float oo  = sigf(o_x + tcc[to + 2*S] + scc[so + 3*S]);
    float nm  = f_ * m[i] + i_ * g_;
    float nc  = c2[i] + ii * gg;
    new_c[i] = nc; new_m[i] = nm; o_out[i] = oo;
    const size_t co = ((size_t)b * 2 * NHC + ch) * HW + hw;
    cell[co] = nc; cell[co + S] = nm;
}

// input scheduling: net = frame (t<10) else mask*frame+(1-mask)*x_gen  (NHWC->NCHW)
__global__ void k_net(const float* __restrict__ frames, const float* __restrict__ mask,
                      const float* __restrict__ xgen, float* __restrict__ net, int t) {
    int i = blockIdx.x * blockDim.x + threadIdx.x;
    if (i >= NB * FCH * HW) return;
    const int hw = i % HW, c = (i / HW) % FCH, b = i / (HW * FCH);
    const float fr = frames[(((size_t)b * TT + t) * HW + hw) * FCH + c];
    float v = fr;
    if (t >= INLEN) {
        int mi = t - INLEN; if (mi > 8) mi = 8;
        const float mk = mask[(((size_t)b * 9 + mi) * HW + hw) * FCH + c];
        v = mk * fr + (1.f - mk) * xgen[i];
    }
    net[i] = v;
}

// emit frame t into d_out ([B][steps][H][W][C]) + accumulate MSE partials
__global__ void k_emit(const float* __restrict__ xgen, const float* __restrict__ frames,
                       float* __restrict__ outp, float* __restrict__ lacc, int t) {
    __shared__ float sred[256];
    const int i = blockIdx.x * 256 + threadIdx.x;
    float d2 = 0.f;
    if (i < NB * FCH * HW) {
        const int hw = i % HW, c = (i / HW) % FCH, b = i / (HW * FCH);
        const float v = xgen[i];
        outp[(((size_t)b * STEPS + t) * HW + hw) * FCH + c] = v;
        const float tg = frames[(((size_t)b * TT + (t + 1)) * HW + hw) * FCH + c];
        const float d = v - tg; d2 = d * d;
    }
    sred[threadIdx.x] = d2;
    __syncthreads();
    for (int off = 128; off > 0; off >>= 1) {
        if ((int)threadIdx.x < off) sred[threadIdx.x] += sred[threadIdx.x + off];
        __syncthreads();
    }
    if (threadIdx.x == 0) atomicAdd(lacc, sred[0]);
}

__global__ void k_finloss(const float* __restrict__ lacc, float* __restrict__ out) {
    out[0] = lacc[0] / (float)((size_t)NB * STEPS * HW * FCH);
}

// ------------------------------ host side ----------------------------------

struct CLn { const float *b, *g, *w; };   // layernorm-conv params (alpha order: b,g,w)

extern "C" void kernel_launch(void* const* d_in, const int* in_sizes, int n_in,
                              void* d_out, int out_size, void* d_ws, size_t ws_size,
                              hipStream_t stream) {
    (void)in_sizes; (void)n_in; (void)out_size; (void)ws_size;
    // ---- parse inputs (JAX pytree flatten: dict keys alphabetical) ----
    int cur = 0;
    const float* frames = (const float*)d_in[cur++];
    const float* mask   = (const float*)d_in[cur++];
    auto getCL = [&]() { CLn c; c.b = (const float*)d_in[cur++];
                         c.g = (const float*)d_in[cur++];
                         c.w = (const float*)d_in[cur++]; return c; };
    // params.cell0: conv_h, conv_last_w, conv_m, conv_o, conv_x
    CLn c0_h = getCL();
    const float* c0_last = (const float*)d_in[cur++];
    CLn c0_m = getCL(); CLn c0_o = getCL(); CLn c0_x = getCL();
    // params.conv_last_w (top-level)
    const float* top_last = (const float*)d_in[cur++];
    // params.diff[3]: conv_h, conv_x, ct_w, oc_w
    struct { CLn h, x; const float *ct, *oc; } df[3];
    for (int j = 0; j < 3; ++j) {
        df[j].h = getCL(); df[j].x = getCL();
        df[j].ct = (const float*)d_in[cur++]; df[j].oc = (const float*)d_in[cur++];
    }
    // params.mim[3]: conv_last_w, ct_w, mims_h, mims_x, oc_w, s_cc, t_cc, x_cc
    struct { const float *last, *ct, *oc; CLn mh, mx, s, t, x; } mm[3];
    for (int j = 0; j < 3; ++j) {
        mm[j].last = (const float*)d_in[cur++];
        mm[j].ct   = (const float*)d_in[cur++];
        mm[j].mh = getCL(); mm[j].mx = getCL();
        mm[j].oc   = (const float*)d_in[cur++];
        mm[j].s = getCL(); mm[j].t = getCL(); mm[j].x = getCL();
    }

    // ---- workspace arena ----
    char* wsp = (char*)d_ws;
    auto alloc = [&](size_t bytes) -> void* {
        void* p = (void*)wsp; wsp += (bytes + 255) & ~(size_t)255; return p;
    };
    const size_t ACT = (size_t)NB * NHC * HW * sizeof(float);   // 2 MB
    float *h0a = (float*)alloc(ACT), *h0b = (float*)alloc(ACT);
    float *hs[4] = { h0a, (float*)alloc(ACT), (float*)alloc(ACT), (float*)alloc(ACT) };
    float *cc[4], *dh[3], *dcb[3], *mcb[3];
    for (int j = 0; j < 4; ++j) cc[j]  = (float*)alloc(ACT);
    for (int j = 0; j < 3; ++j) dh[j]  = (float*)alloc(ACT);
    for (int j = 0; j < 3; ++j) dcb[j] = (float*)alloc(ACT);
    for (int j = 0; j < 3; ++j) mcb[j] = (float*)alloc(ACT);
    float* stmem = (float*)alloc(ACT);
    float* cat2  = (float*)alloc(2 * ACT);
    float* diffb = (float*)alloc(ACT);
    float* xgen  = (float*)alloc((size_t)NB * FCH * HW * sizeof(float));
    float* net   = (float*)alloc((size_t)NB * FCH * HW * sizeof(float));
    float* T448  = (float*)alloc(7 * ACT);
    float* T256a = (float*)alloc(4 * ACT);
    float* T256b = (float*)alloc(4 * ACT);
    float* T256c = (float*)alloc(4 * ACT);
    float* T192  = (float*)alloc(3 * ACT);
    float* T64a  = (float*)alloc(ACT);
    float* T64b  = (float*)alloc(ACT);
    _Float16* Btbuf = (_Float16*)alloc((size_t)NPIX * 3200 * sizeof(_Float16));
    float* lacc = (float*)alloc(256);

    // ---- fp16 weight matrices (padded K) ----
    auto wpad = [](int K) { return (K + 31) & ~31; };
    auto mkW = [&](const float* W, int Cout, int K) -> _Float16* {
        const int Kp = wpad(K);
        _Float16* A = (_Float16*)alloc((size_t)Cout * Kp * sizeof(_Float16));
        const int total = Cout * Kp;
        k_wcast<<<(total + 255) / 256, 256, 0, stream>>>(W, A, K, Kp, total);
        return A;
    };
    _Float16 *Ac0x = mkW(c0_x.w, 7 * NHC, FCH * 25);
    _Float16 *Ac0h = mkW(c0_h.w, 4 * NHC, NHC * 25);
    _Float16 *Ac0m = mkW(c0_m.w, 3 * NHC, NHC * 25);
    _Float16 *Ac0o = mkW(c0_o.w, NHC, 2 * NHC * 25);
    _Float16 *Ac0l = mkW(c0_last, NHC, 2 * NHC);
    _Float16 *Atop = mkW(top_last, FCH, NHC);
    _Float16 *Adh[3], *Adx[3], *At[3], *As_[3], *Ax[3], *Amh[3], *Amx[3], *Al[3];
    for (int j = 0; j < 3; ++j) {
        Adh[j] = mkW(df[j].h.w, 4 * NHC, NHC * 25);
        Adx[j] = mkW(df[j].x.w, 4 * NHC, NHC * 25);
        At[j]  = mkW(mm[j].t.w, 3 * NHC, NHC * 25);
        As_[j] = mkW(mm[j].s.w, 4 * NHC, NHC * 25);
        Ax[j]  = mkW(mm[j].x.w, 4 * NHC, NHC * 25);
        Amh[j] = mkW(mm[j].mh.w, 4 * NHC, NHC * 25);
        Amx[j] = mkW(mm[j].mx.w, 4 * NHC, NHC * 25);
        Al[j]  = mkW(mm[j].last, NHC, 2 * NHC);
    }

    // ---- zero initial state ----
    const int NE = NB * NHC * HW;
    auto zero = [&](float* p, int n) { k_zero<<<(n + 255) / 256, 256, 0, stream>>>(p, n); };
    zero(h0a, NE); zero(hs[1], NE); zero(hs[2], NE); zero(hs[3], NE);
    for (int j = 0; j < 4; ++j) zero(cc[j], NE);
    for (int j = 0; j < 3; ++j) { zero(dh[j], NE); zero(dcb[j], NE); zero(mcb[j], NE); }
    zero(stmem, NE); zero(xgen, NB * FCH * HW); zero(lacc, 1);

    // ---- conv helpers ----
    auto gemm = [&](const _Float16* A, const _Float16* Bt, float* out, int Kp, int cout) {
        k_gemm<<<dim3(NPIX / 128, (cout + 31) / 32), 256, 0, stream>>>(A, Bt, out, Kp, cout);
    };
    auto conv5 = [&](const float* X, const _Float16* A, int cin, int cout, float* out) {
        const int K = cin * 25, Kp = wpad(K);
        const size_t tot = (size_t)NPIX * Kp;
        k_im2col<<<(int)((tot + 255) / 256), 256, 0, stream>>>(X, Btbuf, cin, K, Kp);
        gemm(A, Btbuf, out, Kp, cout);
    };
    auto conv_ln = [&](const float* X, const _Float16* A, const CLn& p,
                       int cin, int cout, float* out) {
        conv5(X, A, cin, cout, out);
        k_layernorm<<<NB, 512, 0, stream>>>(out, p.g, p.b, out, cout);
    };
    auto conv1 = [&](const float* X, const _Float16* A, int cin, int cout, float* out) {
        const size_t tot = (size_t)NPIX * cin;
        k_xpose<<<(int)((tot + 255) / 256), 256, 0, stream>>>(X, Btbuf, cin);
        gemm(A, Btbuf, out, cin, cout);
    };
    const int GE = (NE + 255) / 256;             // elementwise grid (NH channels)
    const int GF = (NB * FCH * HW + 255) / 256;  // elementwise grid (FC channels)

    float* h0cur = h0a;
    float* h0nxt = h0b;

    // ---- time loop ----
    for (int t = 0; t < STEPS; ++t) {
        k_net<<<GF, 256, 0, stream>>>(frames, mask, xgen, net, t);

        // --- SpatioTemporalLSTM cell0 ---
        conv_ln(net,   Ac0x, c0_x, FCH,     7 * NHC, T448);
        conv_ln(h0cur, Ac0h, c0_h, NHC,     4 * NHC, T256a);
        conv_ln(stmem, Ac0m, c0_m, NHC,     3 * NHC, T192);
        k_stlstm_gates<<<GE, 256, 0, stream>>>(T448, T256a, T192, cc[0], stmem,
                                               cc[0], stmem, cat2);
        conv_ln(cat2, Ac0o, c0_o, 2 * NHC, NHC, T64a);
        conv1(cat2, Ac0l, 2 * NHC, NHC, T64b);
        k_stlstm_out<<<GE, 256, 0, stream>>>(T448, T256a, T64a, T64b, h0nxt);
        k_sub<<<GE, 256, 0, stream>>>(h0nxt, h0cur, diffb, NE);
        std::swap(h0cur, h0nxt);
        hs[0] = h0cur;

        // --- MIM layers 1..3 ---
        for (int j = 0; j < 3; ++j) {
            const float* xin = (j == 0) ? diffb : dh[j - 1];
            if (t > 0) {   // diff cell (outputs discarded at t==0 in reference)
                conv_ln(dh[j], Adh[j], df[j].h, NHC, 4 * NHC, T256a);
                conv_ln(xin,   Adx[j], df[j].x, NHC, 4 * NHC, T256b);
                k_mimn<<<GE, 256, 0, stream>>>(T256a, T256b, dcb[j], df[j].ct,
                                               df[j].oc, dh[j], dcb[j]);
            }
            float* hcur  = hs[j + 1];
            const float* hprev = hs[j];
            conv_ln(hcur,      At[j],  mm[j].t,  NHC, 3 * NHC, T192);
            conv_ln(stmem,     As_[j], mm[j].s,  NHC, 4 * NHC, T256a);
            conv_ln(hprev,     Ax[j],  mm[j].x,  NHC, 4 * NHC, T256b);
            conv_ln(cc[j + 1], Amh[j], mm[j].mh, NHC, 4 * NHC, T256c);
            conv_ln(dh[j],     Amx[j], mm[j].mx, NHC, 4 * NHC, T448);
            k_mimn<<<GE, 256, 0, stream>>>(T256c, T448, mcb[j], mm[j].ct, mm[j].oc,
                                           T64a /*c2*/, mcb[j]);
            k_mimblock_gates<<<GE, 256, 0, stream>>>(T192, T256a, T256b, stmem, T64a,
                                                     cc[j + 1], stmem, T64b, cat2);
            conv1(cat2, Al[j], 2 * NHC, NHC, T64a);
            k_mul_tanh<<<GE, 256, 0, stream>>>(T64b, T64a, hcur, NE);
        }

        // --- frame generation + loss accumulation ---
        conv1(hs[3], Atop, NHC, FCH, xgen);
        k_emit<<<GF, 256, 0, stream>>>(xgen, frames, (float*)d_out, lacc, t);
    }

    k_finloss<<<1, 1, 0, stream>>>(lacc, (float*)d_out + (size_t)NB * STEPS * HW * FCH);
}
